// RPA_37658273251505
// MI455X (gfx1250) — matrix operand order repaired and checked by reading, not compiled, not run on hardware
//
#include <hip/hip_runtime.h>
#include <hip/hip_bf16.h>

typedef __attribute__((ext_vector_type(16))) _Float16 v16h;
typedef __attribute__((ext_vector_type(8)))  _Float16 v8h;
typedef __attribute__((ext_vector_type(8)))  float    v8f;

union Frag16 { v16h v; v8h h[2]; };

// A-matrix fragment (16-bit A 16x32 layout): lane holds row M=lane&15,
// K chunks at {b8..b8+7} and {b8+16..b8+23}, b8 = (lane>=16)?8:0.
__device__ __forceinline__ v16h load_fragA(const _Float16* p) {
  Frag16 f;
  f.h[0] = *(const v8h*)(p);
  f.h[1] = *(const v8h*)(p + 16);
  return f.v;
}

// B-matrix fragment (16-bit B 32x16): lane holds col N=lane&15,
// K chunks at {b8..b8+7} and {b8+16..b8+23}, b8 = (lane>=16)?8:0.
__device__ __forceinline__ v16h load_fragB(const _Float16* p) {
  Frag16 f;
  f.h[0] = *(const v8h*)(p);
  f.h[1] = *(const v8h*)(p + 16);
  return f.v;
}

__device__ __forceinline__ v8f wmma_f16(v16h a, v16h b, v8f c) {
  return __builtin_amdgcn_wmma_f32_16x16x32_f16(false, a, false, b,
                                                (short)0, c, false, false);
}

// ---------------------------------------------------------------- converts
__global__ void k_cvt(const float* __restrict__ s, _Float16* __restrict__ d, int n) {
  int i = blockIdx.x * blockDim.x + threadIdx.x;
  int stride = gridDim.x * blockDim.x;
  for (; i < n; i += stride) d[i] = (_Float16)s[i];
}

// x [n][C][pix] f32 -> xhT [n][pix][C] f16, LDS-tiled transpose (coalesced both sides)
__global__ void k_xpose(const float* __restrict__ x, _Float16* __restrict__ xhT) {
  __shared__ _Float16 tile[32][33];
  int pt = blockIdx.x * 32, ct = blockIdx.y * 32, n = blockIdx.z;
  int c = threadIdx.x & 31, r = threadIdx.x >> 5;       // 256 thr: 8 rows x 32
  for (int rr = r; rr < 32; rr += 8)
    tile[rr][c] = (_Float16)x[((size_t)n * 512 + ct + rr) * 1024 + pt + c];
  __syncthreads();
  for (int rr = r; rr < 32; rr += 8)
    xhT[((size_t)n * 1024 + pt + rr) * 512 + ct + c] = tile[c][rr];
}

// fusion_w [o][cc][r9] f32 -> wh2 [o][r9][cc] f16 (linear K-order r9*384+cc)
__global__ void k_wreorder(const float* __restrict__ fusion_w, _Float16* __restrict__ wh2) {
  int i = blockIdx.x * blockDim.x + threadIdx.x;
  if (i >= 512 * 3456) return;
  int cc = i % 384;
  int r9 = (i / 384) % 9;
  int o  = i / 3456;
  wh2[i] = (_Float16)fusion_w[((size_t)o * 384 + cc) * 9 + r9];
}

// ---------------------------------------------------------------- pooling
__global__ void k_pool(const float* __restrict__ x, float* __restrict__ pooled) {
  int bid = blockIdx.x;                 // n*512 + C
  int t = threadIdx.x;                  // 256
  __shared__ float part[256];
  const float* p = x + (size_t)bid * 1024;
  float lsum = p[t*4] + p[t*4+1] + p[t*4+2] + p[t*4+3];
  part[t] = lsum;
  __syncthreads();
  if (t < 21) {
    float acc = 0.f;
    for (int u = 0; u < 256; ++u) {
      int i = u >> 3;
      int j = (u * 4) & 31;
      bool match;
      if (t == 0)      match = true;
      else if (t < 5)  match = (1 + (i >> 4) * 2 + (j >> 4)) == t;
      else             match = (5 + (i >> 3) * 4 + (j >> 3)) == t;
      if (match) acc += part[u];
    }
    float cnt = (t == 0) ? 1024.f : (t < 5 ? 256.f : 64.f);
    pooled[(size_t)bid * 21 + t] = acc / cnt;
  }
}

// ---------------------------------------------------------------- fold q_w/k_w into rce_w
__global__ void k_w2(const float* __restrict__ rce_w, const float* __restrict__ rce_b,
                     const float* __restrict__ q_w, const float* __restrict__ q_b,
                     const float* __restrict__ k_w, const float* __restrict__ k_b,
                     float* __restrict__ w2, float* __restrict__ biasb) {
  int bid = blockIdx.x;                 // 96 = 3 * 2 * 16
  int s = bid / 32, rem = bid % 32, which = rem >> 4, qc = rem & 15;
  const float* W = which ? k_w : q_w;
  const float* B = which ? k_b : q_b;
  int t = threadIdx.x;                  // 256
  for (int col = t; col < 512; col += 256) {
    float acc = 0.f;
    for (int c = 0; c < 128; ++c)
      acc += W[qc * 128 + c] * rce_w[((size_t)s * 128 + c) * 512 + col];
    w2[(((size_t)s * 2 + which) * 16 + qc) * 512 + col] = acc;
  }
  if (t == 0) {
    float acc = B[qc];
    for (int c = 0; c < 128; ++c) acc += W[qc * 128 + c] * rce_b[s * 128 + c];
    biasb[(s * 2 + which) * 16 + qc] = acc;
  }
}

// bilinear (align_corners=True) interpolation matrix entry M_S[o][a], O=32
__device__ __forceinline__ float mentry(int S, int o, int a) {
  if (S == 1) return (a == 0) ? 1.f : 0.f;
  float pos = (float)o * ((float)(S - 1) / 31.0f);
  int lo = (int)floorf(pos);
  if (lo > S - 2) lo = S - 2;
  float frac = pos - (float)lo;
  if (a == lo)     return 1.f - frac;
  if (a == lo + 1) return frac;
  return 0.f;
}

// ---------------------------------------------------------------- q/k generation
// qhT/khT layout: [s][n][pixel(1024)][ch(32)] f16, ch 16..31 zero padded
__global__ void k_qk(const float* __restrict__ pooled, const float* __restrict__ w2,
                     const float* __restrict__ biasb,
                     _Float16* __restrict__ qhT, _Float16* __restrict__ khT) {
  int bid = blockIdx.x;                 // 48 = 16 n * 3 s
  int n = bid / 3, s = bid % 3;
  const int szs[3]   = {1, 2, 4};
  const int cbase[3] = {0, 1, 5};
  int sz = szs[s], sz2 = sz * sz;
  __shared__ float g[2 * 16 * 16];
  int t = threadIdx.x;                  // 256
  int entries = 2 * 16 * sz2;
  if (t < entries) {
    int which = t / (16 * sz2);
    int r = t % (16 * sz2);
    int qc = r / sz2, cell = r % sz2;
    const float* wrow = w2 + (((size_t)s * 2 + which) * 16 + qc) * 512;
    float acc = 0.f;
    for (int C = 0; C < 512; ++C)
      acc += wrow[C] * pooled[((size_t)n * 512 + C) * 21 + cbase[s] + cell];
    g[(which * 16 + qc) * 16 + cell] = acc;
  }
  __syncthreads();
  for (int p = t; p < 1024; p += 256) {
    int i = p >> 5, j = p & 31;
    float Mi[4], Mj[4];
    for (int a = 0; a < sz; ++a) { Mi[a] = mentry(sz, i, a); Mj[a] = mentry(sz, j, a); }
    for (int which = 0; which < 2; ++which) {
      _Float16* dst = (which ? khT : qhT) + (((size_t)s * 16 + n) * 1024 + p) * 32;
      float tmp[16];
      for (int qc = 0; qc < 16; ++qc) {
        float v = biasb[(s * 2 + which) * 16 + qc];
        for (int a = 0; a < sz; ++a)
          for (int b = 0; b < sz; ++b)
            v += Mi[a] * Mj[b] * g[(which * 16 + qc) * 16 + a * sz + b];
        tmp[qc] = v;
      }
      v8h h0, h1, z = {};
      for (int qc = 0; qc < 8; ++qc) { h0[qc] = (_Float16)tmp[qc]; h1[qc] = (_Float16)tmp[qc + 8]; }
      v8h* d = (v8h*)dst;
      d[0] = h0; d[1] = h1; d[2] = z; d[3] = z;
    }
  }
}

// ---------------------------------------------------------------- value projection GEMM
// Vh[n][c4(128)][L(1024)] f16 = f16( value_w(128x512) @ x(512x1024) + value_b )
// block 128 thr (4 waves), tile M=128 x N=64, K-step 32, double-buffered LDS
__global__ void k_gemm_value(const _Float16* __restrict__ vwh, const _Float16* __restrict__ xhT,
                             const float* __restrict__ vb, _Float16* __restrict__ Vh) {
  int Nbase = blockIdx.x * 64;          // pixel tile
  int n = blockIdx.z;
  __shared__ _Float16 Ah[2][128 * 32];  // [c4row][K32]
  __shared__ _Float16 Bh[2][64 * 32];   // [pix][K32]
  int t = threadIdx.x, wave = t >> 5, ln = t & 31;
  int b8 = (ln >= 16) ? 8 : 0, kb = (ln >= 16) ? 16 : 0;
  int ncol = ln & 15, mhi = (ln >> 4) * 8;
  int brow = t >> 1, bpart = t & 1;

  auto stage = [&](int ks, int buf) {
    const v8h* sA = (const v8h*)(vwh + (size_t)t * 512 + ks * 32);
    v8h* dA = (v8h*)(Ah[buf] + t * 32);
    dA[0] = sA[0]; dA[1] = sA[1]; dA[2] = sA[2]; dA[3] = sA[3];
    const v8h* sB = (const v8h*)(xhT + ((size_t)n * 1024 + Nbase + brow) * 512 + ks * 32 + bpart * 16);
    v8h* dB = (v8h*)(Bh[buf] + brow * 32 + bpart * 16);
    dB[0] = sB[0]; dB[1] = sB[1];
  };

  v8f zero = {};
  v8f acc[2][4];
  for (int i = 0; i < 2; ++i)
    for (int j = 0; j < 4; ++j) acc[i][j] = zero;

  stage(0, 0);
  __syncthreads();
  for (int ks = 0; ks < 16; ++ks) {
    int buf = ks & 1;
    if (ks + 1 < 16) stage(ks + 1, buf ^ 1);
    v16h a0 = load_fragA(Ah[buf] + (wave * 32 + ncol) * 32 + b8);
    v16h a1 = load_fragA(Ah[buf] + (wave * 32 + 16 + ncol) * 32 + b8);
    for (int j = 0; j < 4; ++j) {
      v16h b = load_fragB(Bh[buf] + (j * 16 + ncol) * 32 + b8);
      acc[0][j] = wmma_f16(a0, b, acc[0][j]);
      acc[1][j] = wmma_f16(a1, b, acc[1][j]);
    }
    __syncthreads();
  }
  for (int i = 0; i < 2; ++i)
    for (int j = 0; j < 4; ++j)
      for (int r = 0; r < 8; ++r) {
        int c4r = wave * 32 + i * 16 + mhi + r;
        int pix = Nbase + j * 16 + ncol;
        Vh[((size_t)n * 128 + c4r) * 1024 + pix] = (_Float16)(acc[i][j][r] + vb[c4r]);
      }
}

// ---------------------------------------------------------------- fused attention
// per WG: one (n, s, 16-row m-tile): S = Q^T K (WMMA) -> softmax -> O = A V^T (WMMA)
// dynamic LDS: S f32[16][1024] | attn f16[16][1024] | kT f16[1024][32] | red f32[128]
__global__ void k_attn(const _Float16* __restrict__ qhT, const _Float16* __restrict__ khT,
                       const _Float16* __restrict__ Vh, _Float16* __restrict__ cat_hT) {
  int mt = blockIdx.x, s = blockIdx.y, n = blockIdx.z;
  int m0 = mt * 16;
  extern __shared__ char smem[];
  float*    Sbuf = (float*)smem;                              // 65536 B
  _Float16* attn = (_Float16*)(smem + 65536);                 // 32768 B
  _Float16* kT   = (_Float16*)(smem + 65536 + 32768);         // 65536 B
  float*    red  = (float*)(smem + 65536 + 32768 + 65536);    // 512 B
  int t = threadIdx.x, wave = t >> 5, ln = t & 31;
  int b8 = (ln >= 16) ? 8 : 0, kb = (ln >= 16) ? 16 : 0;
  int ncol = ln & 15, mhi = (ln >> 4) * 8;

  {   // stage k slice (1024 x 32ch f16, 64KB) via async global->LDS (ASYNCcnt)
    unsigned long long sbase =
        (unsigned long long)(khT + ((size_t)s * 16 + n) * 1024 * 32);
    unsigned ldsbase = (unsigned)(uintptr_t)kT;
    for (int i = t; i < 4096; i += 128) {
      unsigned voff = (unsigned)i * 16u;
      unsigned ldst = ldsbase + voff;
      asm volatile("global_load_async_to_lds_b128 %0, %1, %2"
                   :: "v"(ldst), "v"(voff), "s"(sbase) : "memory");
    }
  }
  // overlap: load Q A-fragment from global while async copy is in flight
  const _Float16* qbase = qhT + (((size_t)s * 16 + n) * 1024 + m0) * 32;
  v16h afrag = load_fragA(qbase + ncol * 32 + b8);
  asm volatile("s_wait_asynccnt 0x0" ::: "memory");
  __syncthreads();

  // ---- QK: S[m][l] = sum_ch q[ch][m]*k[ch][l], K=32 (padded)
  for (int lt = wave; lt < 64; lt += 4) {
    v16h bfrag = load_fragB(kT + (lt * 16 + ncol) * 32 + b8);
    v8f c = {};
    c = wmma_f16(afrag, bfrag, c);
    for (int r = 0; r < 8; ++r)
      Sbuf[(mhi + r) * 1024 + lt * 16 + ncol] = c[r];
  }
  __syncthreads();

  // ---- softmax rows (16 rows x 1024), 8 segments of 128 per row
  {
    int rowi = t >> 3, seg = t & 7;
    float* sp = Sbuf + rowi * 1024 + seg * 128;
    float mx = -3.4e38f;
    for (int c = 0; c < 128; ++c) mx = fmaxf(mx, sp[c]);
    red[rowi * 8 + seg] = mx;
    __syncthreads();
    float rmax = red[rowi * 8];
    for (int c = 1; c < 8; ++c) rmax = fmaxf(rmax, red[rowi * 8 + c]);
    __syncthreads();
    float sm = 0.f;
    for (int c = 0; c < 128; ++c) { float e = __expf(sp[c] - rmax); sp[c] = e; sm += e; }
    red[rowi * 8 + seg] = sm;
    __syncthreads();
    float rsum = 0.f;
    for (int c = 0; c < 8; ++c) rsum += red[rowi * 8 + c];
    float inv = 1.f / rsum;
    _Float16* ap = attn + rowi * 1024 + seg * 128;
    for (int c8 = 0; c8 < 16; ++c8) {       // vectorized f16 writeback
      v8h hv;
      for (int u = 0; u < 8; ++u) hv[u] = (_Float16)(sp[c8 * 8 + u] * inv);
      *(v8h*)(ap + c8 * 8) = hv;
    }
  }
  __syncthreads();

  // ---- PV: O[m][c] = sum_l attn[m][l] * V[c][l]; each wave owns 32 c-channels
  v8f oc0 = {}, oc1 = {};
  const _Float16* vbase = Vh + (size_t)n * 128 * 1024;
  for (int kt = 0; kt < 32; ++kt) {
    v16h a = load_fragA(attn + ncol * 1024 + kt * 32 + b8);
    int c0 = wave * 32 + ncol;
    v16h bA = load_fragB(vbase + (size_t)c0 * 1024 + kt * 32 + b8);
    oc0 = wmma_f16(a, bA, oc0);
    v16h bB = load_fragB(vbase + (size_t)(c0 + 16) * 1024 + kt * 32 + b8);
    oc1 = wmma_f16(a, bB, oc1);
  }
  // epilogue -> cat_hT [n][pix][cc(384)] f16 (transposed for fusion-conv staging)
  for (int r = 0; r < 8; ++r) {
    int pix = m0 + mhi + r;
    int cc0 = s * 128 + wave * 32 + ncol;
    cat_hT[((size_t)n * 1024 + pix) * 384 + cc0]      = (_Float16)oc0[r];
    cat_hT[((size_t)n * 1024 + pix) * 384 + cc0 + 16] = (_Float16)oc1[r];
  }
}

// ---------------------------------------------------------------- fusion 3x3 conv
// implicit GEMM, K order = r9*384 + cc (matches wh2), tile M=128 x N=64,
// double-buffered LDS; B staging = shifted-row vector copies from cat_hT.
__global__ void k_gemm_fusion(const _Float16* __restrict__ wh2, const _Float16* __restrict__ cat_hT,
                              float* __restrict__ y) {
  int Nbase = blockIdx.x * 64;          // pixel tile
  int Mbase = blockIdx.y * 128;         // output-channel tile
  int n = blockIdx.z;
  __shared__ _Float16 Ah[2][128 * 32];  // [o][K32]
  __shared__ _Float16 Bh[2][64 * 32];   // [pix][K32]
  int t = threadIdx.x, wave = t >> 5, ln = t & 31;
  int b8 = (ln >= 16) ? 8 : 0, kb = (ln >= 16) ? 16 : 0;
  int ncol = ln & 15, mhi = (ln >> 4) * 8;
  int brow = t >> 1, bpart = t & 1;
  int pix = Nbase + brow;
  int py = pix >> 5, px = pix & 31;

  auto stage = [&](int ks, int buf) {
    const v8h* sA = (const v8h*)(wh2 + (size_t)(Mbase + t) * 3456 + ks * 32);
    v8h* dA = (v8h*)(Ah[buf] + t * 32);
    dA[0] = sA[0]; dA[1] = sA[1]; dA[2] = sA[2]; dA[3] = sA[3];
    int r9 = ks / 12, ccb = (ks % 12) * 32;
    int dy = r9 / 3 - 1, dx = r9 % 3 - 1;
    int qy = py + dy, qx = px + dx;
    v8h* dB = (v8h*)(Bh[buf] + brow * 32 + bpart * 16);
    if (qy >= 0 && qy < 32 && qx >= 0 && qx < 32) {
      const v8h* sB = (const v8h*)(cat_hT + ((size_t)n * 1024 + qy * 32 + qx) * 384 + ccb + bpart * 16);
      dB[0] = sB[0]; dB[1] = sB[1];
    } else {
      v8h z = {};
      dB[0] = z; dB[1] = z;
    }
  };

  v8f zero = {};
  v8f acc[2][4];
  for (int i = 0; i < 2; ++i)
    for (int j = 0; j < 4; ++j) acc[i][j] = zero;

  stage(0, 0);
  __syncthreads();
  for (int ks = 0; ks < 108; ++ks) {    // 3456 / 32
    int buf = ks & 1;
    if (ks + 1 < 108) stage(ks + 1, buf ^ 1);
    v16h a0 = load_fragA(Ah[buf] + (wave * 32 + ncol) * 32 + b8);
    v16h a1 = load_fragA(Ah[buf] + (wave * 32 + 16 + ncol) * 32 + b8);
    for (int j = 0; j < 4; ++j) {
      v16h b = load_fragB(Bh[buf] + (j * 16 + ncol) * 32 + b8);
      acc[0][j] = wmma_f16(a0, b, acc[0][j]);
      acc[1][j] = wmma_f16(a1, b, acc[1][j]);
    }
    __syncthreads();
  }
  for (int i = 0; i < 2; ++i)
    for (int j = 0; j < 4; ++j)
      for (int r = 0; r < 8; ++r) {
        int o  = Mbase + wave * 32 + i * 16 + mhi + r;
        int op = Nbase + j * 16 + ncol;
        y[((size_t)n * 512 + o) * 1024 + op] = acc[i][j][r];
      }
}

// ---------------------------------------------------------------- BN batch stats
__global__ void k_bnstats(const float* __restrict__ y, const float* __restrict__ bn_scale,
                          float* __restrict__ bnp) {
  int o = blockIdx.x, t = threadIdx.x;
  __shared__ float ssum[256], ssq[256];
  float s = 0.f, q = 0.f;
  for (int i = t; i < 16 * 1024; i += 256) {
    float v = y[(((size_t)(i >> 10) * 512 + o) << 10) + (i & 1023)];
    s += v; q += v * v;
  }
  ssum[t] = s; ssq[t] = q;
  __syncthreads();
  for (int st = 128; st > 0; st >>= 1) {
    if (t < st) { ssum[t] += ssum[t + st]; ssq[t] += ssq[t + st]; }
    __syncthreads();
  }
  if (t == 0) {
    float mean = ssum[0] / 16384.f;
    float var = ssq[0] / 16384.f - mean * mean;
    bnp[o] = mean;
    bnp[512 + o] = bn_scale[o] * rsqrtf(var + 1e-5f);
  }
}

// ---------------------------------------------------------------- BN apply + ReLU + residual
__global__ void k_final4(const float4* __restrict__ y, const float4* __restrict__ x,
                         const float* __restrict__ bnp, const float* __restrict__ bn_bias,
                         const float* __restrict__ gamma, float4* __restrict__ out, int total4) {
  int i = blockIdx.x * blockDim.x + threadIdx.x;
  if (i >= total4) return;
  int o = (i >> 8) & 511;               // 4 consecutive floats share one channel
  float mean = bnp[o], scl = bnp[512 + o], bb = bn_bias[o], g = gamma[0];
  float4 v = y[i], xx = x[i], r;
  r.x = g * fmaxf((v.x - mean) * scl + bb, 0.f) + xx.x;
  r.y = g * fmaxf((v.y - mean) * scl + bb, 0.f) + xx.y;
  r.z = g * fmaxf((v.z - mean) * scl + bb, 0.f) + xx.z;
  r.w = g * fmaxf((v.w - mean) * scl + bb, 0.f) + xx.w;
  out[i] = r;
}

// ================================================================ launcher
extern "C" void kernel_launch(void* const* d_in, const int* in_sizes, int n_in,
                              void* d_out, int out_size, void* d_ws, size_t ws_size,
                              hipStream_t stream) {
  const float* x        = (const float*)d_in[0];
  const float* rce_w    = (const float*)d_in[1];
  const float* rce_b    = (const float*)d_in[2];
  const float* q_w      = (const float*)d_in[3];
  const float* q_b      = (const float*)d_in[4];
  const float* k_w      = (const float*)d_in[5];
  const float* k_b      = (const float*)d_in[6];
  const float* value_w  = (const float*)d_in[7];
  const float* value_b  = (const float*)d_in[8];
  const float* fusion_w = (const float*)d_in[9];
  const float* bn_scale = (const float*)d_in[10];
  const float* bn_bias  = (const float*)d_in[11];
  const float* gamma    = (const float*)d_in[12];
  float* out = (float*)d_out;

  char* ws = (char*)d_ws;
  size_t off = 0;
  auto take = [&](size_t bytes) -> void* {
    void* p = ws + off;
    off = (off + bytes + 255) & ~(size_t)255;
    return p;
  };
  _Float16* xhT  = (_Float16*)take((size_t)16 * 1024 * 512 * 2);
  _Float16* vwh  = (_Float16*)take((size_t)128 * 512 * 2);
  _Float16* wh2  = (_Float16*)take((size_t)512 * 3456 * 2);
  _Float16* Vh   = (_Float16*)take((size_t)16 * 128 * 1024 * 2);
  float*    pooled = (float*)take((size_t)16 * 512 * 21 * 4);
  float*    w2     = (float*)take((size_t)3 * 2 * 16 * 512 * 4);
  float*    biasb  = (float*)take((size_t)3 * 2 * 16 * 4);
  _Float16* qhT  = (_Float16*)take((size_t)3 * 16 * 1024 * 32 * 2);
  _Float16* khT  = (_Float16*)take((size_t)3 * 16 * 1024 * 32 * 2);
  _Float16* catT = (_Float16*)take((size_t)16 * 1024 * 384 * 2);
  float*    ybuf = (float*)take((size_t)16 * 512 * 1024 * 4);
  float*    bnp  = (float*)take((size_t)512 * 2 * 4);
  (void)ws_size; (void)in_sizes; (void)n_in; (void)out_size;

  // operand prep: transpose/convert x, convert value_w, reorder+convert fusion_w
  k_xpose<<<dim3(32, 16, 16), 256, 0, stream>>>(x, xhT);
  k_cvt<<<256, 256, 0, stream>>>(value_w, vwh, 128 * 512);
  k_wreorder<<<(512 * 3456 + 255) / 256, 256, 0, stream>>>(fusion_w, wh2);

  // pyramid pooling + folded q/k weights + q/k generation
  k_pool<<<16 * 512, 256, 0, stream>>>(x, pooled);
  k_w2<<<96, 256, 0, stream>>>(rce_w, rce_b, q_w, q_b, k_w, k_b, w2, biasb);
  k_qk<<<48, 256, 0, stream>>>(pooled, w2, biasb, qhT, khT);

  // value projection (WMMA, M=128 whole, double-buffered)
  k_gemm_value<<<dim3(16, 1, 16), 128, 0, stream>>>(vwh, xhT, value_b, Vh);

  // fused attention per (n, scale, m-tile) (async-to-LDS + WMMA + softmax)
  k_attn<<<dim3(64, 3, 16), 128, 164608, stream>>>(qhT, khT, Vh, catT);

  // fusion 3x3 conv as implicit GEMM (WMMA, double-buffered)
  k_gemm_fusion<<<dim3(16, 4, 16), 128, 0, stream>>>(wh2, catT, ybuf);

  // batchnorm (training-mode batch stats) + relu + residual
  k_bnstats<<<512, 256, 0, stream>>>(ybuf, bn_scale, bnp);
  int total4 = 16 * 512 * 1024 / 4;
  k_final4<<<(total4 + 255) / 256, 256, 0, stream>>>((const float4*)ybuf, (const float4*)x,
                                                     bnp, bn_bias, gamma, (float4*)out, total4);
}
